// TemporalBranch_38998303047991
// MI455X (gfx1250) — matrix operand order repaired and checked by reading, not compiled, not run on hardware
//
#include <hip/hip_runtime.h>
#include <math.h>

// ---------- feature detection: CDNA5 async global->LDS copies ----------
#if defined(__has_builtin)
#  if __has_builtin(__builtin_amdgcn_global_load_async_to_lds_b128) && \
      __has_builtin(__builtin_amdgcn_s_wait_asynccnt)
#    define USE_ASYNC_LDS 1
#  endif
#endif
#ifndef USE_ASYNC_LDS
#  define USE_ASYNC_LDS 0
#endif

// ---------- types ----------
typedef __attribute__((ext_vector_type(16))) __bf16 v16bf;
typedef __attribute__((ext_vector_type(8)))  float  v8f;

union FragU { uint4 q[2]; v16bf v; };

__device__ __forceinline__ unsigned short f2bf(float f) {
  union { float f; unsigned u; } c; c.f = f;
  unsigned u = c.u;
  unsigned r = u + 0x7FFFu + ((u >> 16) & 1u);   // round-to-nearest-even
  return (unsigned short)(r >> 16);
}

#if USE_ASYNC_LDS
// builtin signature (from hipcc diagnostic): param0 is
// 'int __attribute__((vector_size(16))) addrspace(1)*' (printed as __device__)
typedef int v4i_ __attribute__((vector_size(4 * sizeof(int))));
typedef __attribute__((address_space(1))) v4i_ as1_v4i;
typedef __attribute__((address_space(3))) v4i_ as3_v4i;
__device__ __forceinline__ void async_ld16(const unsigned short* g, unsigned short* l) {
  // flat LDS address bits [31:0] == LDS offset (ISA 10.2 aperture mapping)
  __builtin_amdgcn_global_load_async_to_lds_b128(
      (as1_v4i*)(unsigned long long)g,
      (as3_v4i*)(unsigned int)(unsigned long long)l,
      0, 0);
}
#endif

// ---------- problem constants ----------
#define BN_SER   112
#define NPATCH   64
#define DLLM     768
#define DMODEL   1024
#define HK       1024
#define NTOK     1000
#define NTOKP    1024          // K-padded token count
#define VOCAB    30522
#define VOCABP   30528         // K-padded vocab (mult of 64)

#define TF_OFF    0L
#define MIN_OFF   5505024L
#define MAX_OFF   5505136L
#define MED_OFF   5505248L
#define TREND_OFF 5505360L
#define LAG_OFF   5505472L
#define NOUT_OFF  5506032L

// =====================================================================
// Batched bf16 GEMM: C = alpha*A(MxK)*B(KxN)+bias.
// A: row-major (M x K), K-contiguous, ldA % 8 == 0.
// Bt: B stored transposed, (N x K), K-contiguous, ldB % 8 == 0.
// K % 64 == 0 (caller zero-pads). M/N may be ragged (clamped staging,
// guarded writeback). Tile 128x128x64, 8 waves x (2x4 tiles x 2 ksteps).
// Staging: async global->LDS (double buffered) when available, else
// register double-buffer + ds_store_b128.
// =====================================================================
#define BM 128
#define BNT 128
#define BK 64
#define LTILE (BM * BK)        // elements per LDS tile buffer

#define INIT_STAGE(R, aP, bP, aS, bS)                                     \
  do {                                                                    \
    int idx = (R) * 256 + tid;                                            \
    int mm = idx >> 3, kv = (idx & 7) * 8;                                \
    int gm = m0 + mm; gm = (gm < M) ? gm : (M - 1);                       \
    int gn = n0 + mm; gn = (gn < N) ? gn : (N - 1);                       \
    aP = A  + (long)gm * ldA + kv;                                        \
    bP = Bt + (long)gn * ldB + kv;                                        \
    aS = &LA0[mm][kv];                                                    \
    bS = &LB0[mm][kv];                                                    \
  } while (0)

__global__ __launch_bounds__(256, 1) void k_gemm(
    const unsigned short* __restrict__ A, long ldA, long aHi, long aLo,
    const unsigned short* __restrict__ Bt, long ldB, long bHi, long bLo,
    void* __restrict__ Cv, long ldcR, long ldcC, long cHi, long cLo, int outBF16,
    int M, int N, int K, int innerB,
    float alpha, const float* __restrict__ bias, int biasMode)
{
#if USE_ASYNC_LDS
  __shared__ __align__(16) unsigned short lA[2][BM][BK];
  __shared__ __align__(16) unsigned short lB[2][BNT][BK];
#define LA0 (lA[0])
#define LB0 (lB[0])
#else
  __shared__ __align__(16) unsigned short lA[BM][BK];
  __shared__ __align__(16) unsigned short lB[BNT][BK];
#define LA0 lA
#define LB0 lB
#endif

  int z = blockIdx.z;
  long zq = z / innerB, zr = z % innerB;
  A  += zq * aHi + zr * aLo;
  Bt += zq * bHi + zr * bLo;
  long coff = zq * cHi + zr * cLo;

  int m0 = blockIdx.y * BM, n0 = blockIdx.x * BNT;
  int tid = threadIdx.x, lane = tid & 31, wave = tid >> 5;
  int wm = wave & 3, wn = wave >> 2;

  const unsigned short *aP0, *aP1, *aP2, *aP3, *bP0, *bP1, *bP2, *bP3;
  unsigned short *aS0, *aS1, *aS2, *aS3, *bS0, *bS1, *bS2, *bS3;
  INIT_STAGE(0, aP0, bP0, aS0, bS0);
  INIT_STAGE(1, aP1, bP1, aS1, bS1);
  INIT_STAGE(2, aP2, bP2, aS2, bS2);
  INIT_STAGE(3, aP3, bP3, aS3, bS3);

  v8f acc[2][4] = {};

  int aRow = lane & 15;
  int aKb  = (lane >> 4) << 3;            // 0 or 8
  int bCol = lane & 15;
  int bKb  = (lane >> 4) << 3;            // 0 or 8

  // one K-tile of WMMAs from the given LDS buffers
  auto mma_tile = [&](const unsigned short (*cA)[BK], const unsigned short (*cB)[BK]) {
    #pragma unroll
    for (int s = 0; s < 2; ++s) {
      FragU fa0, fa1;
      int m = wm * 32 + aRow;
      fa0.q[0] = *(const uint4*)&cA[m][s * 32 + aKb];
      fa0.q[1] = *(const uint4*)&cA[m][s * 32 + 16 + aKb];
      fa1.q[0] = *(const uint4*)&cA[m + 16][s * 32 + aKb];
      fa1.q[1] = *(const uint4*)&cA[m + 16][s * 32 + 16 + aKb];
      #pragma unroll
      for (int ni = 0; ni < 4; ++ni) {
        FragU fb;
        int n = wn * 64 + ni * 16 + bCol;
        fb.q[0] = *(const uint4*)&cB[n][s * 32 + bKb];
        fb.q[1] = *(const uint4*)&cB[n][s * 32 + 16 + bKb];
        acc[0][ni] = __builtin_amdgcn_wmma_f32_16x16x32_bf16(
            false, fa0.v, false, fb.v, (short)0, acc[0][ni], false, false);
        acc[1][ni] = __builtin_amdgcn_wmma_f32_16x16x32_bf16(
            false, fa1.v, false, fb.v, (short)0, acc[1][ni], false, false);
      }
    }
  };

#if USE_ASYNC_LDS
  // ---- async double-buffered staging (ASYNCcnt-tracked DMA) ----
  async_ld16(aP0, aS0); async_ld16(aP1, aS1);
  async_ld16(aP2, aS2); async_ld16(aP3, aS3);
  async_ld16(bP0, bS0); async_ld16(bP1, bS1);
  async_ld16(bP2, bS2); async_ld16(bP3, bS3);

  int it = 0;
  for (int k0 = 0; k0 < K; k0 += BK, ++it) {
    __builtin_amdgcn_s_wait_asynccnt(0);   // this wave's tile landed in LDS
    __syncthreads();                       // all waves landed; prev buffer free
    int kn = k0 + BK;
    if (kn < K) {
      __builtin_prefetch((const void*)(aP0 + kn + BK), 0, 3);
      __builtin_prefetch((const void*)(bP0 + kn + BK), 0, 3);
      long o = (long)((it + 1) & 1) * LTILE;
      async_ld16(aP0 + kn, aS0 + o); async_ld16(aP1 + kn, aS1 + o);
      async_ld16(aP2 + kn, aS2 + o); async_ld16(aP3 + kn, aS3 + o);
      async_ld16(bP0 + kn, bS0 + o); async_ld16(bP1 + kn, bS1 + o);
      async_ld16(bP2 + kn, bS2 + o); async_ld16(bP3 + kn, bS3 + o);
    }
    mma_tile(lA[it & 1], lB[it & 1]);
  }
#else
  // ---- register double-buffer + ds_store staging ----
  uint4 pa0 = *(const uint4*)aP0, pa1 = *(const uint4*)aP1;
  uint4 pa2 = *(const uint4*)aP2, pa3 = *(const uint4*)aP3;
  uint4 pb0 = *(const uint4*)bP0, pb1 = *(const uint4*)bP1;
  uint4 pb2 = *(const uint4*)bP2, pb3 = *(const uint4*)bP3;

  for (int k0 = 0; k0 < K; k0 += BK) {
    *(uint4*)aS0 = pa0; *(uint4*)aS1 = pa1;
    *(uint4*)aS2 = pa2; *(uint4*)aS3 = pa3;
    *(uint4*)bS0 = pb0; *(uint4*)bS1 = pb1;
    *(uint4*)bS2 = pb2; *(uint4*)bS3 = pb3;
    __syncthreads();

    int kn = k0 + BK;
    if (kn < K) {
      __builtin_prefetch((const void*)(aP0 + kn + BK), 0, 3);
      __builtin_prefetch((const void*)(bP0 + kn + BK), 0, 3);
      pa0 = *(const uint4*)(aP0 + kn); pa1 = *(const uint4*)(aP1 + kn);
      pa2 = *(const uint4*)(aP2 + kn); pa3 = *(const uint4*)(aP3 + kn);
      pb0 = *(const uint4*)(bP0 + kn); pb1 = *(const uint4*)(bP1 + kn);
      pb2 = *(const uint4*)(bP2 + kn); pb3 = *(const uint4*)(bP3 + kn);
    }
    mma_tile(LA0, LB0);
    __syncthreads();
  }
#endif
#undef LA0
#undef LB0

  // ---- writeback ----
  int cCol = lane & 15, rBase = (lane >> 4) << 3;
  float* Cf = (float*)Cv;
  unsigned short* Cb = (unsigned short*)Cv;
  #pragma unroll
  for (int mi = 0; mi < 2; ++mi) {
    #pragma unroll
    for (int ni = 0; ni < 4; ++ni) {
      int col = n0 + wn * 64 + ni * 16 + cCol;
      if (col >= N) continue;
      #pragma unroll
      for (int r = 0; r < 8; ++r) {
        int row = m0 + wm * 32 + mi * 16 + rBase + r;
        if (row >= M) continue;
        float val = acc[mi][ni][r] * alpha;
        if (biasMode == 1) val += bias[col];
        else if (biasMode == 2) val += bias[row];
        long ci = coff + (long)row * ldcR + (long)col * ldcC;
        if (outBF16) Cb[ci] = f2bf(val); else Cf[ci] = val;
      }
    }
  }
}

// =====================================================================
// Tiled transpose+convert: src (R x C) f32 -> dst (C x Rpad) bf16,
// rows r in [R, Rpad) zero-filled (K padding must be exact zero).
// =====================================================================
__global__ void k_transpose_cvt(const float* __restrict__ src, unsigned short* __restrict__ dst,
                                int R, int C, int Rpad) {
  __shared__ float tile[32][33];
  int c0 = blockIdx.x * 32, r0 = blockIdx.y * 32;
  int tx = threadIdx.x & 31, ty = threadIdx.x >> 5;
  #pragma unroll
  for (int i = 0; i < 32; i += 8) {
    int r = r0 + ty + i, c = c0 + tx;
    tile[ty + i][tx] = (r < R && c < C) ? src[(long)r * C + c] : 0.0f;
  }
  __syncthreads();
  #pragma unroll
  for (int i = 0; i < 32; i += 8) {
    int c = c0 + ty + i, r = r0 + tx;
    if (c < C && r < Rpad) dst[(long)c * Rpad + r] = f2bf(tile[tx][ty + i]);
  }
}

// =====================================================================
// Patch matrix X (7168 x 64) bf16, cols 48..63 zero (K pad)
// =====================================================================
__global__ void k_buildX(const float* __restrict__ xe, unsigned short* __restrict__ X) {
  long i = (long)blockIdx.x * 256 + threadIdx.x;
  if (i >= (long)BN_SER * NPATCH * 64) return;
  int c = (int)(i & 63), row = (int)(i >> 6);
  if (c >= 48) { X[i] = 0; return; }
  int bn = row >> 6, l = row & 63;
  int kk = c >> 4, ii = c & 15;
  int col = (l - 1 + kk + 64) & 63;
  int t = col * 8 + ii; if (t > 511) t = 511;
  int b = bn / 7, n = bn % 7;
  X[i] = f2bf(xe[((long)(b * 512 + t)) * 7 + n]);
}

// =====================================================================
// Conv weight reorder -> Wf (48 x 1024) bf16: Wf[kk*16+i][o] = w[o][i][kk]
// =====================================================================
__global__ void k_buildWf(const float* __restrict__ w, unsigned short* __restrict__ Wf) {
  long i = (long)blockIdx.x * 256 + threadIdx.x;
  if (i >= 48L * 1024) return;
  int o = (int)(i % 1024), r = (int)(i / 1024);
  int ii = r & 15, kk = r >> 4;
  Wf[i] = f2bf(w[(long)o * 48 + ii * 3 + kk]);
}

// =====================================================================
// Per-series stats: min/max/median (bitonic) + trend + N output
// =====================================================================
__global__ void k_stats(const float* __restrict__ xe, float* __restrict__ out) {
  int s = blockIdx.x;
  int b = s / 7, n = s % 7;
  __shared__ float v[512];
  int tid = threadIdx.x;
  for (int t = tid; t < 512; t += 256) v[t] = xe[((long)(b * 512 + t)) * 7 + n];
  __syncthreads();
  if (tid == 0) out[TREND_OFF + s] = v[511] - v[0];
  __syncthreads();
  for (int ksz = 2; ksz <= 512; ksz <<= 1) {
    for (int j = ksz >> 1; j > 0; j >>= 1) {
      for (int i = tid; i < 512; i += 256) {
        int ixj = i ^ j;
        if (ixj > i) {
          float a = v[i], c = v[ixj];
          bool up = ((i & ksz) == 0);
          if (up ? (a > c) : (a < c)) { v[i] = c; v[ixj] = a; }
        }
      }
      __syncthreads();
    }
  }
  if (tid == 0) {
    out[MIN_OFF + s] = v[0];
    out[MAX_OFF + s] = v[511];
    out[MED_OFF + s] = v[255];
    if (s == 0) out[NOUT_OFF] = 7.0f;
  }
}

// =====================================================================
// Circular autocorrelation + top-5 lags
// =====================================================================
__global__ void k_autocorr(const float* __restrict__ xe, float* __restrict__ out) {
  int s = blockIdx.x;
  int b = s / 7, n = s % 7;
  __shared__ float x[512];
  __shared__ float corr[512];
  __shared__ float sv[256];
  __shared__ int   si[256];
  int tid = threadIdx.x;
  for (int t = tid; t < 512; t += 256) x[t] = xe[((long)(b * 512 + t)) * 7 + n];
  __syncthreads();
  for (int lag = tid; lag < 512; lag += 256) {
    float acc = 0.0f;
    for (int t = 0; t < 512; ++t) acc += x[t] * x[(t + lag) & 511];
    corr[lag] = acc;
  }
  __syncthreads();
  for (int it = 0; it < 5; ++it) {
    float bv = -3.4e38f; int bi = 0;
    for (int lag = tid; lag < 512; lag += 256) {
      float c = corr[lag];
      if (c > bv || (c == bv && lag < bi)) { bv = c; bi = lag; }
    }
    sv[tid] = bv; si[tid] = bi;
    __syncthreads();
    for (int st = 128; st > 0; st >>= 1) {
      if (tid < st) {
        if (sv[tid + st] > sv[tid] ||
            (sv[tid + st] == sv[tid] && si[tid + st] < si[tid])) {
          sv[tid] = sv[tid + st]; si[tid] = si[tid + st];
        }
      }
      __syncthreads();
    }
    if (tid == 0) { out[LAG_OFF + (long)s * 5 + it] = (float)si[0]; corr[si[0]] = -3.4e38f; }
    __syncthreads();
  }
}

// =====================================================================
// Row softmax over 1000 scores -> bf16 probs, row stride 1024, tail zeroed
// =====================================================================
__global__ void k_softmax(const float* __restrict__ sc, unsigned short* __restrict__ pr) {
  long row = blockIdx.x;
  const float* srow = sc + row * NTOK;
  unsigned short* prow = pr + row * NTOKP;
  __shared__ float red[256];
  int tid = threadIdx.x;
  float m = -3.4e38f;
  for (int i = tid; i < NTOK; i += 256) m = fmaxf(m, srow[i]);
  red[tid] = m; __syncthreads();
  for (int st = 128; st > 0; st >>= 1) { if (tid < st) red[tid] = fmaxf(red[tid], red[tid + st]); __syncthreads(); }
  m = red[0]; __syncthreads();
  float ssum = 0.0f;
  for (int i = tid; i < NTOK; i += 256) ssum += __expf(srow[i] - m);
  red[tid] = ssum; __syncthreads();
  for (int st = 128; st > 0; st >>= 1) { if (tid < st) red[tid] += red[tid + st]; __syncthreads(); }
  float inv = 1.0f / red[0];
  for (int i = tid; i < NTOKP; i += 256)
    prow[i] = (i < NTOK) ? f2bf(__expf(srow[i] - m) * inv) : (unsigned short)0;
}

// =====================================================================
// host side
// =====================================================================
static inline void gemm_launch(hipStream_t s,
    const unsigned short* A, long ldA, long aHi, long aLo,
    const unsigned short* Bt, long ldB, long bHi, long bLo,
    void* C, long ldcR, long ldcC, long cHi, long cLo, int outBF16,
    int M, int N, int K, int batch, int innerB,
    float alpha, const float* bias, int biasMode)
{
  dim3 g((N + BNT - 1) / BNT, (M + BM - 1) / BM, batch);
  k_gemm<<<g, 256, 0, s>>>(A, ldA, aHi, aLo, Bt, ldB, bHi, bLo,
                           C, ldcR, ldcC, cHi, cLo, outBF16, M, N, K, innerB,
                           alpha, bias, biasMode);
}

static inline void tr_launch(hipStream_t s, const float* src, unsigned short* dst,
                             int R, int C, int Rpad) {
  k_transpose_cvt<<<dim3((C + 31) / 32, (Rpad + 31) / 32), 256, 0, s>>>(src, dst, R, C, Rpad);
}

extern "C" void kernel_launch(void* const* d_in, const int* in_sizes, int n_in,
                              void* d_out, int out_size, void* d_ws, size_t ws_size,
                              hipStream_t stream) {
  (void)in_sizes; (void)n_in; (void)out_size; (void)ws_size;
  const float* x_enc = (const float*)d_in[0];
  const float* We    = (const float*)d_in[1];
  const float* map_w = (const float*)d_in[2];
  const float* map_b = (const float*)d_in[3];
  const float* pconv = (const float*)d_in[4];
  const float* wq    = (const float*)d_in[5];
  const float* bq    = (const float*)d_in[6];
  const float* wk    = (const float*)d_in[7];
  const float* bk    = (const float*)d_in[8];
  const float* wv    = (const float*)d_in[9];
  const float* bv    = (const float*)d_in[10];
  const float* wo    = (const float*)d_in[11];
  const float* bo    = (const float*)d_in[12];
  float* out = (float*)d_out;

  size_t off = 0;
  auto bump = [&](size_t bytes) -> char* {
    char* p = (char*)d_ws + off;
    off += (bytes + 255) & ~(size_t)255;
    return p;
  };
  unsigned short* WeBt  = (unsigned short*)bump((size_t)DLLM * VOCABP * 2);   // (768 x 30528)
  unsigned short* mapT  = (unsigned short*)bump((size_t)NTOK * VOCABP * 2);   // (1000 x 30528)
  unsigned short* srcB  = (unsigned short*)bump((size_t)NTOK * DLLM * 2);     // (1000 x 768)
  unsigned short* wqBt  = (unsigned short*)bump((size_t)HK * DMODEL * 2);     // (1024 x 1024)
  unsigned short* wkBt  = (unsigned short*)bump((size_t)HK * DLLM * 2);       // (1024 x 768)
  unsigned short* wvBt  = (unsigned short*)bump((size_t)HK * DLLM * 2);       // (1024 x 768)
  unsigned short* woBt  = (unsigned short*)bump((size_t)DLLM * HK * 2);       // (768 x 1024)
  unsigned short* WfB   = (unsigned short*)bump(48UL * DMODEL * 2);           // (48 x 1024)
  unsigned short* W2Bt  = (unsigned short*)bump((size_t)HK * 64 * 2);         // (1024 x 64)
  unsigned short* XB    = (unsigned short*)bump((size_t)BN_SER * NPATCH * 64 * 2); // (7168 x 64)
  unsigned short* qB    = (unsigned short*)bump((size_t)BN_SER * NPATCH * HK * 2); // (7168 x 1024)
  unsigned short* kB    = (unsigned short*)bump((size_t)NTOK * HK * 2);       // (1000 x 1024)
  unsigned short* vBt   = (unsigned short*)bump((size_t)HK * NTOKP * 2);      // (1024 x 1024)
  float*          scF   = (float*)bump((size_t)BN_SER * 8 * NPATCH * NTOK * 4);
  unsigned short* prB   = (unsigned short*)bump((size_t)BN_SER * 8 * NPATCH * NTOKP * 2);
  unsigned short* attnB = (unsigned short*)bump((size_t)BN_SER * NPATCH * HK * 2);

  // ---- transposed bf16 weight/operand prep (K pads zero-filled) ----
  tr_launch(stream, We,    WeBt, VOCAB, DLLM,  VOCABP);
  tr_launch(stream, map_w, mapT, VOCAB, NTOK,  VOCABP);
  tr_launch(stream, wq,    wqBt, DMODEL, HK,   DMODEL);
  tr_launch(stream, wk,    wkBt, DLLM,  HK,    DLLM);
  tr_launch(stream, wv,    wvBt, DLLM,  HK,    DLLM);
  tr_launch(stream, wo,    woBt, HK,    DLLM,  HK);
  k_buildWf<<<dim3((48 * 1024 + 255) / 256), 256, 0, stream>>>(pconv, WfB);
  k_buildX<<<dim3(((long)BN_SER * NPATCH * 64 + 255) / 256), 256, 0, stream>>>(x_enc, XB);

  // ---- statistics branch ----
  k_stats<<<dim3(BN_SER), 256, 0, stream>>>(x_enc, out);
  k_autocorr<<<dim3(BN_SER), 256, 0, stream>>>(x_enc, out);

  // ---- src = map_wT @ We + map_b (per-row bias) -> bf16 (1000x768) ----
  gemm_launch(stream, mapT, VOCABP, 0, 0,
              WeBt, VOCABP, 0, 0,
              srcB, DLLM, 1, 0, 0, 1,
              NTOK, DLLM, VOCABP, 1, 1, 1.0f, map_b, 2);

  // ---- W2 = Wf @ wq, written transposed (1024 x 64) ----
  gemm_launch(stream, WfB, DMODEL, 0, 0,
              wqBt, DMODEL, 0, 0,
              W2Bt, 1, 64, 0, 0, 1,
              48, HK, DMODEL, 1, 1, 1.0f, nullptr, 0);

  // ---- q = X @ W2 + bq -> bf16 (7168 x 1024) ----
  gemm_launch(stream, XB, 64, 0, 0,
              W2Bt, 64, 0, 0,
              qB, HK, 1, 0, 0, 1,
              BN_SER * NPATCH, HK, 64, 1, 1, 1.0f, bq, 1);

  // ---- k = src @ wk + bk (natural); v = src @ wv + bv (transposed) ----
  gemm_launch(stream, srcB, DLLM, 0, 0,
              wkBt, DLLM, 0, 0,
              kB, HK, 1, 0, 0, 1,
              NTOK, HK, DLLM, 1, 1, 1.0f, bk, 1);
  gemm_launch(stream, srcB, DLLM, 0, 0,
              wvBt, DLLM, 0, 0,
              vBt, 1, NTOKP, 0, 0, 1,
              NTOK, HK, DLLM, 1, 1, 1.0f, bv, 1);

  // ---- scores[b,h] = scale * Q(64x128) @ K_h^T ; Bt == kB natural ----
  const float scale = 0.08838834764831845f;
  gemm_launch(stream,
              qB, HK, (long)NPATCH * HK, 128,
              kB, HK, 0, 128,
              scF, NTOK, 1, (long)8 * NPATCH * NTOK, (long)NPATCH * NTOK, 0,
              NPATCH, NTOK, 128, BN_SER * 8, 8, scale, nullptr, 0);

  // ---- softmax rows -> bf16 probs (ld 1024, K-pad zeroed) ----
  k_softmax<<<dim3(BN_SER * 8 * NPATCH), 256, 0, stream>>>(scF, prB);

  // ---- attn[b,h] = P(64x1024p) @ V_h ; Bt == vBt slice ----
  gemm_launch(stream,
              prB, NTOKP, (long)8 * NPATCH * NTOKP, (long)NPATCH * NTOKP,
              vBt, NTOKP, 0, (long)128 * NTOKP,
              attnB, HK, 1, (long)NPATCH * HK, 128, 1,
              NPATCH, 128, NTOKP, BN_SER * 8, 8, 1.0f, nullptr, 0);

  // ---- temporal_features = attn @ wo + bo -> d_out f32 ----
  gemm_launch(stream, attnB, HK, 0, 0,
              woBt, HK, 0, 0,
              out + TF_OFF, DLLM, 1, 0, 0, 0,
              BN_SER * NPATCH, DLLM, HK, 1, 1, 1.0f, bo, 1);
}
